// ConvAttention_12240656793864
// MI455X (gfx1250) — compile-verified, matrix operand order checked by reading, not run on hardware
//
#include <hip/hip_runtime.h>
#include <hip/hip_bf16.h>

typedef __attribute__((ext_vector_type(16))) _Float16 v16h;
typedef __attribute__((ext_vector_type(8)))  _Float16 v8h;
typedef __attribute__((ext_vector_type(8)))  float    v8f;

#define BB 2
#define TT 4096
#define DD 256
#define HH 8
#define DKK 32
#define EPSV 1e-5f

// ---------------------------------------------------------------------------
// A/B fragment loader for v_wmma_f32_16x16x32_f16.
// 16-bit A-matrix 16x32 layout (ISA 7.12.2): lane half g holds
//   elements 0..7  -> K = g*8 + e
//   elements 8..15 -> K = 16 + g*8 + e
// p points at (row_base + k_base); works for row-major global or LDS rows.
// ---------------------------------------------------------------------------
__device__ __forceinline__ v16h load_frag16(const _Float16* p, int g) {
    v8h lo = *(const v8h*)(p + g * 8);
    v8h hi = *(const v8h*)(p + 16 + g * 8);
    v16h r;
#pragma unroll
    for (int e = 0; e < 8; ++e) { r[e] = lo[e]; r[8 + e] = hi[e]; }
    return r;
}

__device__ __forceinline__ v8f wmma_f16(v16h a, v16h b, v8f c) {
    return __builtin_amdgcn_wmma_f32_16x16x32_f16(false, a, false, b,
                                                  (short)0, c, false, false);
}

// ---------------------------------------------------------------------------
// Kernel: f32 -> f16 conversion (weights)
// ---------------------------------------------------------------------------
__global__ void ca_f32_to_f16(const float* __restrict__ src,
                              _Float16* __restrict__ dst, int n) {
    int idx = blockIdx.x * 256 + threadIdx.x;
    if (idx < n) dst[idx] = (_Float16)src[idx];
}

// ---------------------------------------------------------------------------
// Kernel: depthwise conv (K=3, same pad) + folded BN -> f16 y[3, B*T, D]
// ---------------------------------------------------------------------------
__global__ void ca_dwconv_bn(const float* __restrict__ x,      // [B,T,D]
                             const float* __restrict__ dww,    // [3,D,1,3]
                             const float* __restrict__ dwb,    // [3,D]
                             const float* __restrict__ bng,
                             const float* __restrict__ bnb,
                             const float* __restrict__ bnm,
                             const float* __restrict__ bnv,
                             _Float16* __restrict__ y) {
    size_t idx = (size_t)blockIdx.x * 256 + threadIdx.x;
    const size_t total = (size_t)3 * BB * TT * DD;
    if (idx >= total) return;
    int c = (int)(idx % DD);
    size_t tmp = idx / DD;
    int t = (int)(tmp % TT);
    tmp /= TT;
    int b = (int)(tmp % BB);
    int i = (int)(tmp / BB);

    const float* xr = x + (size_t)b * TT * DD + c;
    int wbase = (i * DD + c) * 3;
    float w0 = dww[wbase + 0], w1 = dww[wbase + 1], w2 = dww[wbase + 2];

    float acc = xr[(size_t)t * DD] * w1;
    if (t > 0)      acc += xr[(size_t)(t - 1) * DD] * w0;
    if (t < TT - 1) acc += xr[(size_t)(t + 1) * DD] * w2;

    int pc = i * DD + c;
    float s = bng[pc] * rsqrtf(bnv[pc] + EPSV);
    float o = (acc + dwb[pc] - bnm[pc]) * s + bnb[pc];
    y[idx] = (_Float16)o;
}

// ---------------------------------------------------------------------------
// Kernel: pointwise GEMM  qkv[i] = y[i] (8192x256) * W[i]^T (256x256) + b[i]
// 128 threads = 4 waves; block covers 16 rows x 256 cols; each wave 16x64.
// Branches 0/1 (Q,K) stored row-major [B*T, D]; branch 2 (V) stored
// transposed as vT[B, D, T] so attention's P*V B-fragments are contiguous.
// ---------------------------------------------------------------------------
__global__ void __launch_bounds__(128)
ca_gemm_qkv(const _Float16* __restrict__ Y,     // [3, B*T, D]
            const _Float16* __restrict__ W,     // [3, D(out), D(in)] f16
            const float* __restrict__ bias,     // [3, D]
            _Float16* __restrict__ qkv,         // [2, B*T, D]  (Q, K)
            _Float16* __restrict__ vT) {        // [B, D, T]    (V transposed)
    int i    = blockIdx.z;
    int mt   = blockIdx.x;                 // 16-row tile index
    int wave = threadIdx.x >> 5;
    int lane = threadIdx.x & 31;
    int rl   = lane & 15;
    int g    = lane >> 4;

    const _Float16* A  = Y + (size_t)i * (BB * TT) * DD + (size_t)mt * 16 * DD;
    const _Float16* Wb = W + (size_t)i * DD * DD;

    v8f acc[4];
#pragma unroll
    for (int j = 0; j < 4; ++j) acc[j] = (v8f){};

    for (int kk = 0; kk < DD; kk += 32) {
        v16h a = load_frag16(A + (size_t)rl * DD + kk, g);
#pragma unroll
        for (int j = 0; j < 4; ++j) {
            int o0 = (wave * 4 + j) * 16;
            v16h bf = load_frag16(Wb + (size_t)(o0 + rl) * DD + kk, g);
            acc[j] = wmma_f16(a, bf, acc[j]);
        }
    }

    if (i < 2) {
        _Float16* outp = qkv + (size_t)i * (BB * TT) * DD + (size_t)mt * 16 * DD;
#pragma unroll
        for (int j = 0; j < 4; ++j) {
            int n = (wave * 4 + j) * 16 + rl;
            float bs = bias[i * DD + n];
#pragma unroll
            for (int r = 0; r < 8; ++r) {
                int M = r + g * 8;
                outp[(size_t)M * DD + n] = (_Float16)(acc[j][r] + bs);
            }
        }
    } else {
        // transposed store: global row (mt*16 + M) -> (b, t); column n -> channel.
        // Lane's 8 acc elements map to t = t0 + 8g + r (contiguous) -> one b128 store.
        int b  = (mt * 16) / TT;
        int t0 = (mt * 16) % TT;
#pragma unroll
        for (int j = 0; j < 4; ++j) {
            int n = (wave * 4 + j) * 16 + rl;
            float bs = bias[2 * DD + n];
            v8h pv;
#pragma unroll
            for (int r = 0; r < 8; ++r) pv[r] = (_Float16)(acc[j][r] + bs);
            *(v8h*)(vT + ((size_t)b * DD + n) * TT + t0 + g * 8) = pv;
        }
    }
}

// ---------------------------------------------------------------------------
// Kernel: flash attention. One wave per (b, h, 16-query tile).
// S = Q(16x32) * Kblk^T  -> 2x wmma (32 keys per block, K-dim = DK = 32)
// online softmax, P through LDS (C-layout -> A-layout), O += P * Vblk.
// V is pre-transposed, so V B-fragments are contiguous b128 loads.
// ---------------------------------------------------------------------------
__global__ void __launch_bounds__(32)
ca_flash_attn(const _Float16* __restrict__ qkv,   // [2, B*T, D] (Q, K)
              const _Float16* __restrict__ vT,    // [B, D, T]
              _Float16* __restrict__ ctx) {       // [B*T, D]
    __shared__ _Float16 plds[16 * 32];

    int qt = blockIdx.x;
    int h  = blockIdx.y;
    int b  = blockIdx.z;
    int lane = threadIdx.x & 31;
    int rl = lane & 15;
    int g  = lane >> 4;

    const _Float16* Q  = qkv + ((size_t)(0 * BB + b) * TT) * DD + (size_t)h * DKK;
    const _Float16* Kp = qkv + ((size_t)(1 * BB + b) * TT) * DD + (size_t)h * DKK;
    // V row for this lane's column: vf0 -> dk = rl, vf1 -> dk = 16 + rl
    const _Float16* V0 = vT + ((size_t)b * DD + h * DKK + rl) * TT;
    const _Float16* V1 = vT + ((size_t)b * DD + h * DKK + 16 + rl) * TT;

    v16h qf = load_frag16(Q + (size_t)(qt * 16 + rl) * DD, g);

    float mrow[8], lrow[8];
#pragma unroll
    for (int r = 0; r < 8; ++r) { mrow[r] = -1e30f; lrow[r] = 0.0f; }
    v8f o0 = (v8f){}, o1 = (v8f){};
    const float scale = 0.17677669529663687f;   // 1/sqrt(32)

    for (int kb = 0; kb < TT / 32; ++kb) {
        int key0 = kb * 32;

        v16h kf0 = load_frag16(Kp + (size_t)(key0 + rl) * DD, g);
        v16h kf1 = load_frag16(Kp + (size_t)(key0 + 16 + rl) * DD, g);

        v8f z = (v8f){};
        v8f s0 = wmma_f16(qf, kf0, z);
        v8f s1 = wmma_f16(qf, kf1, z);

        // V block fragments: contiguous along key axis thanks to vT layout.
        v16h vf0 = load_frag16(V0 + key0, g);
        v16h vf1 = load_frag16(V1 + key0, g);

        // online softmax per row (rows striped: VGPR r <-> M = r + 8*g)
#pragma unroll
        for (int r = 0; r < 8; ++r) {
            float a0 = s0[r] * scale;
            float a1 = s1[r] * scale;
            float mx = fmaxf(a0, a1);
#pragma unroll
            for (int off = 1; off < 16; off <<= 1)
                mx = fmaxf(mx, __shfl_xor(mx, off, 32));
            float nm = fmaxf(mrow[r], mx);
            float alpha = __expf(mrow[r] - nm);
            float p0 = __expf(a0 - nm);
            float p1 = __expf(a1 - nm);
            float rs = p0 + p1;
#pragma unroll
            for (int off = 1; off < 16; off <<= 1)
                rs += __shfl_xor(rs, off, 32);
            lrow[r] = lrow[r] * alpha + rs;
            mrow[r] = nm;
            o0[r] *= alpha;
            o1[r] *= alpha;
            int M = r + g * 8;
            plds[M * 32 + rl]      = (_Float16)p0;
            plds[M * 32 + 16 + rl] = (_Float16)p1;
        }

        asm volatile("s_wait_dscnt 0x0" ::: "memory");
        v16h pf = load_frag16(&plds[rl * 32], g);
        asm volatile("s_wait_dscnt 0x0" ::: "memory");

        o0 = wmma_f16(pf, vf0, o0);
        o1 = wmma_f16(pf, vf1, o1);
    }

    _Float16* Cp = ctx + ((size_t)b * TT) * DD + (size_t)h * DKK;
#pragma unroll
    for (int r = 0; r < 8; ++r) {
        int M = r + g * 8;
        float inv = 1.0f / lrow[r];
        Cp[(size_t)(qt * 16 + M) * DD + rl]      = (_Float16)(o0[r] * inv);
        Cp[(size_t)(qt * 16 + M) * DD + 16 + rl] = (_Float16)(o1[r] * inv);
    }
}

// ---------------------------------------------------------------------------
// Kernel: output projection  out = ctx (8192x256) * out_w^T + out_b  (f32 out)
// ---------------------------------------------------------------------------
__global__ void __launch_bounds__(128)
ca_gemm_out(const _Float16* __restrict__ Cx,    // [B*T, D]
            const _Float16* __restrict__ W,     // [D(out), D(in)] f16
            const float* __restrict__ bias,     // [D]
            float* __restrict__ out) {          // [B*T, D] f32
    int mt   = blockIdx.x;
    int wave = threadIdx.x >> 5;
    int lane = threadIdx.x & 31;
    int rl   = lane & 15;
    int g    = lane >> 4;

    const _Float16* A = Cx + (size_t)mt * 16 * DD;

    v8f acc[4];
#pragma unroll
    for (int j = 0; j < 4; ++j) acc[j] = (v8f){};

    for (int kk = 0; kk < DD; kk += 32) {
        v16h a = load_frag16(A + (size_t)rl * DD + kk, g);
#pragma unroll
        for (int j = 0; j < 4; ++j) {
            int o0 = (wave * 4 + j) * 16;
            v16h bf = load_frag16(W + (size_t)(o0 + rl) * DD + kk, g);
            acc[j] = wmma_f16(a, bf, acc[j]);
        }
    }

    float* outp = out + (size_t)mt * 16 * DD;
#pragma unroll
    for (int j = 0; j < 4; ++j) {
        int n = (wave * 4 + j) * 16 + rl;
        float bs = bias[n];
#pragma unroll
        for (int r = 0; r < 8; ++r) {
            int M = r + g * 8;
            outp[(size_t)M * DD + n] = acc[j][r] + bs;
        }
    }
}

// ---------------------------------------------------------------------------
extern "C" void kernel_launch(void* const* d_in, const int* in_sizes, int n_in,
                              void* d_out, int out_size, void* d_ws, size_t ws_size,
                              hipStream_t stream) {
    const float* x    = (const float*)d_in[0];
    const float* dww  = (const float*)d_in[1];
    const float* dwb  = (const float*)d_in[2];
    const float* bng  = (const float*)d_in[3];
    const float* bnb  = (const float*)d_in[4];
    const float* bnm  = (const float*)d_in[5];
    const float* bnv  = (const float*)d_in[6];
    const float* pww  = (const float*)d_in[7];
    const float* pwb  = (const float*)d_in[8];
    const float* outw = (const float*)d_in[9];
    const float* outb = (const float*)d_in[10];
    float* out = (float*)d_out;

    const size_t n_y   = (size_t)3 * BB * TT * DD;   // 6,291,456 halves
    const size_t n_btd = (size_t)BB * TT * DD;       // 2,097,152 halves
    _Float16* y     = (_Float16*)d_ws;               // [3, B*T, D]
    _Float16* qkv   = y + n_y;                       // [2, B*T, D] (Q, K)
    _Float16* vT    = qkv + 2 * n_btd;               // [B, D, T]
    _Float16* ctx   = vT + n_btd;                    // [B*T, D]
    _Float16* pwwh  = ctx + n_btd;                   // [3, D, D]
    _Float16* outwh = pwwh + (size_t)3 * DD * DD;    // [D, D]

    ca_f32_to_f16<<<(3 * DD * DD + 255) / 256, 256, 0, stream>>>(pww, pwwh, 3 * DD * DD);
    ca_f32_to_f16<<<(DD * DD + 255) / 256, 256, 0, stream>>>(outw, outwh, DD * DD);

    ca_dwconv_bn<<<(unsigned)((n_y + 255) / 256), 256, 0, stream>>>(
        x, dww, dwb, bng, bnb, bnm, bnv, y);

    ca_gemm_qkv<<<dim3(BB * TT / 16, 1, 3), 128, 0, stream>>>(y, pwwh, pwb, qkv, vT);

    ca_flash_attn<<<dim3(TT / 16, HH, BB), 32, 0, stream>>>(qkv, vT, ctx);

    ca_gemm_out<<<dim3(BB * TT / 16, 1, 1), 128, 0, stream>>>(ctx, outwh, outb, out);
}
